// VarianceAdaptor_74680891343270
// MI455X (gfx1250) — matrix operand order repaired and checked by reading, not compiled, not run on hardware
//
#include <hip/hip_runtime.h>
#include <hip/hip_bf16.h>

#define Bb 64
#define Tt 1024
#define Hc 256
#define OUT_LEN 3072           // T * MAX_DUR
#define MT 64                  // output rows (time steps) per block
#define LROWS 66               // MT + 2 halo rows
#define LPAD 264               // halves per LDS row (256 + 8 pad, keeps 16B align)

typedef __attribute__((ext_vector_type(16))) _Float16 v16h;
typedef __attribute__((ext_vector_type(8)))  _Float16 v8h;
typedef __attribute__((ext_vector_type(8)))  float    v8f;

// ---------------------------------------------------------------------------
// f32 -> f16 convert (vectorized x4)
// ---------------------------------------------------------------------------
__global__ void f32_to_f16_kernel(const float* __restrict__ in,
                                  _Float16* __restrict__ out, int n4) {
    int i = blockIdx.x * 256 + threadIdx.x;
    if (i >= n4) return;
    float4 v = *(const float4*)(in + (size_t)i * 4);
    _Float16* o = out + (size_t)i * 4;
    o[0] = (_Float16)v.x; o[1] = (_Float16)v.y;
    o[2] = (_Float16)v.z; o[3] = (_Float16)v.w;
}

// ---------------------------------------------------------------------------
// Pre-swizzle conv weights [O=256][I=256][K=3] (f32) into WMMA B-fragment
// order [k][kstep(8)][ntile(16)][lane(32)][j(16)] f16 (32 contiguous bytes
// per lane).  B layout used here (16-bit, 32x16):
//   lanes 0-15 : N = lane,    K-label = kk0 + j
//   lanes 16-31: N = lane-16, K-label = kk0 + 16 + j
// K relabeling: so that the A operand can be read as 16 *contiguous* halves
// per lane (lane reads positions lhi*16..lhi*16+15, hardware labels them
// kappa = (j<8 ? j : j+8) + 8*lhi), B uses the same positions, q = kl.
// Valid because sum_k A[m,k]B[k,n] is invariant under a K permutation
// applied to both operands; A (x, h) stays in natural channel order.
// ---------------------------------------------------------------------------
__global__ void prep_wfrag_kernel(const float* __restrict__ w,
                                  _Float16* __restrict__ wfrag) {
    int idx = blockIdx.x * 256 + threadIdx.x;       // 3*8*16*32*16 = 196608
    if (idx >= 3 * 8 * 16 * 32 * 16) return;
    int j    = idx & 15;
    int lane = (idx >> 4) & 31;
    int nt   = (idx >> 9) & 15;
    int is   = (idx >> 13) & 7;
    int k    = idx >> 16;
    int n  = nt * 16 + (lane & 15);
    int kl = ((lane >= 16) ? 16 : 0) + j;           // K-label within group
    int q  = kl;
    int kk = is * 32 + q;                           // actual input channel
    wfrag[idx] = (_Float16)w[((size_t)n * Hc + kk) * 3 + k];
}

// ---------------------------------------------------------------------------
// Conv1d(k=3, SAME) + bias + ReLU as WMMA GEMM.
// Block: 256 thr (8 waves) computes 64 rows x 256 cols.  Each wave owns a
// 64x32 sub-tile = 4(M) x 2(N) WMMA 16x16 tiles, so every global B-fragment
// (L2-resident weights) feeds 4 WMMAs; A-fragments come from LDS as a single
// contiguous 32B read per lane (K relabeled, see prep_wfrag_kernel).
// ---------------------------------------------------------------------------
__global__ __launch_bounds__(256)
void conv_wmma_kernel(const _Float16* __restrict__ xin,    // [B,T,H] f16
                      const _Float16* __restrict__ wfrag,  // pre-swizzled
                      const float*    __restrict__ bias,   // [H]
                      _Float16*       __restrict__ yout) { // [B,T,H] f16
    __shared__ _Float16 lds[LROWS * LPAD];

    int blk = blockIdx.x;            // B*T/MT = 1024 blocks
    int b   = blk >> 4;
    int t0  = (blk & 15) * MT;
    int tid = threadIdx.x;

    // Stage rows [t0-1, t0+64] into LDS (zero halo at sequence edges).
    for (int c = tid; c < LROWS * 32; c += 256) {
        int r   = c >> 5;
        int col = (c & 31) * 8;
        int tg  = t0 - 1 + r;
        v8h v = {};
        if (tg >= 0 && tg < Tt)
            v = *(const v8h*)(xin + ((size_t)b * Tt + tg) * Hc + col);
        *(v8h*)(&lds[r * LPAD + col]) = v;
    }
    __syncthreads();

    int wave = tid >> 5, lane = tid & 31;   // wave: N cols [wave*32, +32)
    int lhi = (lane >= 16) ? 1 : 0;
    int l16 = lane & 15;

    v8f acc[4][2];
#pragma unroll
    for (int mt = 0; mt < 4; ++mt)
#pragma unroll
        for (int nt = 0; nt < 2; ++nt) acc[mt][nt] = (v8f){};

    for (int k = 0; k < 3; ++k) {
        for (int is = 0; is < 8; ++is) {
            int i0 = is * 32;
            // A fragments: one contiguous 32B LDS read per lane (relabeled K)
            v16h afrag[4];
#pragma unroll
            for (int mt = 0; mt < 4; ++mt) {
                int lrow = mt * 16 + l16 + k;         // tap shift via +k
                afrag[mt] = *(const v16h*)(&lds[lrow * LPAD + i0 + lhi * 16]);
            }
            // B fragments: contiguous 32B per lane from pre-swizzled weights.
            const _Float16* wb =
                wfrag + ((((size_t)k * 8 + is) * 16 + wave * 2) * 32 + lane) * 16;
#pragma unroll
            for (int nt = 0; nt < 2; ++nt) {
                v16h bfr = *(const v16h*)(wb + (size_t)nt * 32 * 16);
#pragma unroll
                for (int mt = 0; mt < 4; ++mt) {
                    acc[mt][nt] = __builtin_amdgcn_wmma_f32_16x16x32_f16(
                        false, afrag[mt], false, bfr,
                        (short)0, acc[mt][nt], false, false);
                }
            }
        }
    }

    // Epilogue: bias + ReLU, f16 store.  C layout: VGPR r -> M = r + 8*lhi,
    // N = lane%16.
#pragma unroll
    for (int nt = 0; nt < 2; ++nt) {
        int n = wave * 32 + nt * 16 + l16;
        float bv = bias[n];
#pragma unroll
        for (int mt = 0; mt < 4; ++mt) {
#pragma unroll
            for (int r = 0; r < 8; ++r) {
                int mloc = mt * 16 + r + lhi * 8;
                float v = acc[mt][nt][r] + bv;
                v = v > 0.0f ? v : 0.0f;
                yout[((size_t)b * Tt + t0 + mloc) * Hc + n] = (_Float16)v;
            }
        }
    }
}

// ---------------------------------------------------------------------------
// Projection: log_dur[row] = dot(h2[row,:], proj_w) + proj_b.  Wave per row.
// ---------------------------------------------------------------------------
__global__ void proj_kernel(const _Float16* __restrict__ h2,
                            const float* __restrict__ pw,
                            const float* __restrict__ pb,
                            float* __restrict__ out) {
    int gw = (blockIdx.x * blockDim.x + threadIdx.x) >> 5;
    int lane = threadIdx.x & 31;
    if (gw >= Bb * Tt) return;
    v8h v = *(const v8h*)(h2 + (size_t)gw * Hc + lane * 8);
    float s = 0.f;
#pragma unroll
    for (int j = 0; j < 8; ++j) s += (float)v[j] * pw[lane * 8 + j];
#pragma unroll
    for (int off = 16; off > 0; off >>= 1) s += __shfl_xor(s, off, 32);
    if (lane == 0) out[gw] = s + pb[0];
}

// ---------------------------------------------------------------------------
// Inclusive cumsum of durations per batch (Hillis-Steele, 1024 threads).
// ---------------------------------------------------------------------------
__global__ void cumsum_kernel(const int* __restrict__ dur, int* __restrict__ cum) {
    __shared__ int s[2][Tt];
    int b = blockIdx.x, t = threadIdx.x;
    s[0][t] = dur[b * Tt + t];
    __syncthreads();
    int src = 0;
    for (int off = 1; off < Tt; off <<= 1) {
        int v = s[src][t];
        if (t >= off) v += s[src][t - off];
        s[src ^ 1][t] = v;
        src ^= 1;
        __syncthreads();
    }
    cum[b * Tt + t] = s[src][t];
}

// ---------------------------------------------------------------------------
// Length regulator: binary-search cumsum, gather x rows, zero masked frames.
// Block = 256 thr handles 4 output rows; 64 thr x float4 per row.
// ---------------------------------------------------------------------------
__global__ void regulate_kernel(const float* __restrict__ x,
                                const int* __restrict__ cum,
                                float* __restrict__ out) {
    int tid = threadIdx.x;
    int rid = blockIdx.x * 4 + (tid >> 6);     // row id = b*OUT_LEN + pos
    int c4  = tid & 63;
    int b   = rid / OUT_LEN;
    int pos = rid - b * OUT_LEN;
    const int* cb = cum + b * Tt;
    int total = cb[Tt - 1];
    float4 v = make_float4(0.f, 0.f, 0.f, 0.f);
    if (pos < total) {
        int lo = 0, hi = Tt;
        while (lo < hi) {                       // searchsorted(side="right")
            int mid = (lo + hi) >> 1;
            if (cb[mid] <= pos) lo = mid + 1; else hi = mid;
        }
        int idx = lo < Tt ? lo : Tt - 1;
        v = *(const float4*)(x + ((size_t)b * Tt + idx) * Hc + c4 * 4);
    }
    *(float4*)(out + (size_t)rid * Hc + c4 * 4) = v;
}

// ---------------------------------------------------------------------------
extern "C" void kernel_launch(void* const* d_in, const int* in_sizes, int n_in,
                              void* d_out, int out_size, void* d_ws, size_t ws_size,
                              hipStream_t stream) {
    const float* x    = (const float*)d_in[0];
    const int*   dur  = (const int*)d_in[1];
    const float* w1   = (const float*)d_in[2];
    const float* b1   = (const float*)d_in[3];
    const float* w2   = (const float*)d_in[4];
    const float* b2   = (const float*)d_in[5];
    const float* pw   = (const float*)d_in[6];
    const float* pb   = (const float*)d_in[7];

    float* out_reg = (float*)d_out;                               // [B,OUT_LEN,H]
    float* out_dur = (float*)d_out + (size_t)Bb * OUT_LEN * Hc;   // [B,T]

    // Workspace layout (bytes)
    char* ws = (char*)d_ws;
    _Float16* xh     = (_Float16*)(ws);                    // 32 MB, reused as h2
    _Float16* h1h    = (_Float16*)(ws + 33554432);         // 32 MB
    _Float16* w1frag = (_Float16*)(ws + 67108864);         // 384 KB
    _Float16* w2frag = (_Float16*)(ws + 67502080);         // 384 KB
    int*      cum    = (int*)     (ws + 67895296);         // 256 KB
    _Float16* h2h    = xh;                                 // reuse after conv1

    // 1. x -> f16
    int n4 = (Bb * Tt * Hc) / 4;
    f32_to_f16_kernel<<<(n4 + 255) / 256, 256, 0, stream>>>(x, xh, n4);

    // 2. weight fragment swizzle
    prep_wfrag_kernel<<<768, 256, 0, stream>>>(w1, w1frag);
    prep_wfrag_kernel<<<768, 256, 0, stream>>>(w2, w2frag);

    // 3. conv1 + ReLU (WMMA)
    conv_wmma_kernel<<<(Bb * Tt) / MT, 256, 0, stream>>>(xh, w1frag, b1, h1h);

    // 4. conv2 + ReLU (WMMA), output overwrites xh region
    conv_wmma_kernel<<<(Bb * Tt) / MT, 256, 0, stream>>>(h1h, w2frag, b2, h2h);

    // 5. projection -> log_duration_prediction
    proj_kernel<<<(Bb * Tt * 32) / 256, 256, 0, stream>>>(h2h, pw, pb, out_dur);

    // 6. cumsum of durations
    cumsum_kernel<<<Bb, Tt, 0, stream>>>(dur, cum);

    // 7. length regulator gather
    regulate_kernel<<<(Bb * OUT_LEN) / 4, 256, 0, stream>>>(x, cum, out_reg);
}
